// KANAutonomousODEFunc_53283364274892
// MI455X (gfx1250) — matrix operand order repaired and checked by reading, not compiled, not run on hardware
//
#include <hip/hip_runtime.h>
#include <hip/hip_bf16.h>
#include <math.h>

// Problem constants (fixed by the reference)
#define BSZ   2048                 // batch (GEMM M)
#define HDIM  1024                 // hidden (GEMM N, and input features)
#define KBAS  16                   // basis functions per feature
#define KDIM  (HDIM * KBAS)        // GEMM reduction dim = 16384
#define KSTEP 32                   // bf16 WMMA reduction per instruction
#define KB_STEPS (KDIM / KSTEP)    // 512 k-iterations
#define MB_FRAGS (BSZ / 16)        // 128 A-fragment rows
#define NB_FRAGS (HDIM / 16)       // 64  B-fragment cols

typedef __attribute__((ext_vector_type(16))) __bf16 v16bf;
typedef __attribute__((ext_vector_type(8)))  float  v8f;

__device__ __forceinline__ unsigned short f32_to_bf16_rne(float f) {
  unsigned u = __float_as_uint(f);
  u += 0x7FFFu + ((u >> 16) & 1u);      // round-to-nearest-even
  return (unsigned short)(u >> 16);
}

// ---------------------------------------------------------------------------
// Kernel 1: W[o,i,k] f32  ->  bf16 B-matrix fragments (32x16, ISA layout):
//   lanes 0-15 : col = lane,    e<8 -> K=e,   e>=8 -> K=e+8   (K 0..7,16..23)
//   lanes 16-31: col = lane-16, e<8 -> K=e+8, e>=8 -> K=e+16  (K 8..15,24..31)
// Since W[o,i,k] flattens (i,k) as flatK = i*16+k, the per-lane 16 K-values
// are two runs of 8 contiguous f32 (src[0..7], src[16..23]) -> 4x float4 loads per thread.
// One thread per (nb, kb, lane); fragment (nb,kb) = 32 lanes x 32B = 1 KB.
// ---------------------------------------------------------------------------
__global__ void __launch_bounds__(256)
prep_w_frags(const float* __restrict__ W, unsigned short* __restrict__ wfrag) {
  const int tid  = blockIdx.x * blockDim.x + threadIdx.x;
  const int lane = tid & 31;
  const int kb   = (tid >> 5) & (KB_STEPS - 1);
  const int nb   = tid >> 14;                 // / (32*512)
  if (nb >= NB_FRAGS) return;

  const int col   = nb * 16 + (lane & 15);
  const int kbase = kb * KSTEP + ((lane < 16) ? 0 : 8);
  const float* __restrict__ src = W + (size_t)col * KDIM + kbase;

  alignas(16) unsigned short tmp[16];
#pragma unroll
  for (int e = 0; e < 16; ++e) tmp[e] = f32_to_bf16_rne(src[e < 8 ? e : e + 8]);

  uint4* dst = (uint4*)wfrag + (size_t)tid * 2;   // 32 B per lane
  dst[0] = *(const uint4*)&tmp[0];
  dst[1] = *(const uint4*)&tmp[8];
}

// ---------------------------------------------------------------------------
// Kernel 2: s[b,i,k] = tanh(h[b,i]*alpha[k]+beta[k]) -> bf16 A-matrix
// fragments (16x32, ISA layout):
//   lanes 0-15 : row=lane,    e<8 -> K=e,     e>=8 -> K=e+8   (K 0..7,16..23)
//   lanes 16-31: row=lane-16, e<8 -> K=e+8,   e>=8 -> K=e+16  (K 8..15,24..31)
// A 32-wide K block spans exactly 2 input features (16 basis each):
//   elements 0..7  use h0 = h[row, 2*kb],   elements 8..15 use h1,
//   basis index = kofs + (e & 7),  kofs = 0 (lanes<16) or 8 (lanes>=16).
// ---------------------------------------------------------------------------
__global__ void __launch_bounds__(256)
prep_s_frags(const float* __restrict__ h, const float* __restrict__ alpha,
             const float* __restrict__ beta, unsigned short* __restrict__ sfrag) {
  const int tid  = blockIdx.x * blockDim.x + threadIdx.x;
  const int lane = tid & 31;
  const int kb   = (tid >> 5) & (KB_STEPS - 1);
  const int mb   = tid >> 14;
  if (mb >= MB_FRAGS) return;

  const int row  = mb * 16 + (lane & 15);
  const int kofs = (lane < 16) ? 0 : 8;
  const float h0 = h[(size_t)row * HDIM + kb * 2 + 0];
  const float h1 = h[(size_t)row * HDIM + kb * 2 + 1];

  alignas(16) unsigned short tmp[16];
#pragma unroll
  for (int j = 0; j < 8; ++j) {
    const float al = alpha[kofs + j];
    const float be = beta[kofs + j];
    tmp[j]     = f32_to_bf16_rne(tanhf(fmaf(h0, al, be)));
    tmp[8 + j] = f32_to_bf16_rne(tanhf(fmaf(h1, al, be)));
  }

  uint4* dst = (uint4*)sfrag + (size_t)tid * 2;
  dst[0] = *(const uint4*)&tmp[0];
  dst[1] = *(const uint4*)&tmp[8];
}

// ---------------------------------------------------------------------------
// Kernel 3: WMMA GEMM  phi = S(2048x16384) * W'(16384x1024), fused epilogue
//   out = tanh(phi/16)*gain + bias
// Block = 256 threads = 8 waves; WG tile 128(M) x 128(N).
// Waves 4(M) x 2(N); each wave owns 32x64 -> 2x4 v8f accumulators,
// 8 v_wmma_f32_16x16x32_bf16 per k-step against 6 fragment loads.
// NOTE: deliberately NOT hand-pipelined — the scheduler produces in-place
// WMMA accumulation, 4x unroll, s_clause'd b128 loads with descending
// partial s_wait_loadcnt, and stays under 256 VGPRs (2 waves/SIMD).
// ---------------------------------------------------------------------------
__global__ void __launch_bounds__(256)
kan_wmma_gemm(const unsigned short* __restrict__ sfrag,
              const unsigned short* __restrict__ wfrag,
              const float* __restrict__ gain, const float* __restrict__ bias,
              float* __restrict__ out) {
  const int lane = threadIdx.x & 31;
  const int wave = threadIdx.x >> 5;
  const int wm   = wave & 3;          // wave position in M (0..3)
  const int wn   = wave >> 2;         // wave position in N (0..1)
  const int mb0  = blockIdx.y * 8 + wm * 2;   // first of 2 A-frag rows
  const int nb0  = blockIdx.x * 8 + wn * 4;   // first of 4 B-frag cols

  const v16bf* __restrict__ SA = (const v16bf*)sfrag;
  const v16bf* __restrict__ WB = (const v16bf*)wfrag;

  // Per-lane fragment streams; one fragment = 32 lanes * one v16bf (32 B).
  const v16bf* a0p = SA + ((size_t)(mb0 + 0) * KB_STEPS) * 32 + lane;
  const v16bf* a1p = SA + ((size_t)(mb0 + 1) * KB_STEPS) * 32 + lane;
  const v16bf* b0p = WB + ((size_t)(nb0 + 0) * KB_STEPS) * 32 + lane;
  const v16bf* b1p = WB + ((size_t)(nb0 + 1) * KB_STEPS) * 32 + lane;
  const v16bf* b2p = WB + ((size_t)(nb0 + 2) * KB_STEPS) * 32 + lane;
  const v16bf* b3p = WB + ((size_t)(nb0 + 3) * KB_STEPS) * 32 + lane;

  v8f acc[2][4];
#pragma unroll
  for (int i = 0; i < 2; ++i)
#pragma unroll
    for (int j = 0; j < 4; ++j) acc[i][j] = (v8f){0.f,0.f,0.f,0.f,0.f,0.f,0.f,0.f};

  for (int kb = 0; kb < KB_STEPS; ++kb) {
    const size_t o = (size_t)kb * 32;
    // Prefetch A streams a few k-steps ahead into the WGP cache
    // (B streams are shared by all M-blocks and stay L2/L0 hot).
    if (kb + 4 < KB_STEPS) {
      __builtin_prefetch((const void*)(a0p + o + 4 * 32), 0, 1);
      __builtin_prefetch((const void*)(a1p + o + 4 * 32), 0, 1);
    }
    const v16bf a0 = a0p[o];
    const v16bf a1 = a1p[o];
    const v16bf b0 = b0p[o];
    const v16bf b1 = b1p[o];
    const v16bf b2 = b2p[o];
    const v16bf b3 = b3p[o];

    acc[0][0] = __builtin_amdgcn_wmma_f32_16x16x32_bf16(false, a0, false, b0, (short)0, acc[0][0], false, false);
    acc[0][1] = __builtin_amdgcn_wmma_f32_16x16x32_bf16(false, a0, false, b1, (short)0, acc[0][1], false, false);
    acc[0][2] = __builtin_amdgcn_wmma_f32_16x16x32_bf16(false, a0, false, b2, (short)0, acc[0][2], false, false);
    acc[0][3] = __builtin_amdgcn_wmma_f32_16x16x32_bf16(false, a0, false, b3, (short)0, acc[0][3], false, false);
    acc[1][0] = __builtin_amdgcn_wmma_f32_16x16x32_bf16(false, a1, false, b0, (short)0, acc[1][0], false, false);
    acc[1][1] = __builtin_amdgcn_wmma_f32_16x16x32_bf16(false, a1, false, b1, (short)0, acc[1][1], false, false);
    acc[1][2] = __builtin_amdgcn_wmma_f32_16x16x32_bf16(false, a1, false, b2, (short)0, acc[1][2], false, false);
    acc[1][3] = __builtin_amdgcn_wmma_f32_16x16x32_bf16(false, a1, false, b3, (short)0, acc[1][3], false, false);
  }

  // Epilogue: C/D 16x16 f32 layout -> VGPR r: lanes 0-15 row=r, lanes16-31 row=r+8
  const float inv_k = 1.0f / (float)KBAS;
#pragma unroll
  for (int jn = 0; jn < 4; ++jn) {
    const int col = (nb0 + jn) * 16 + (lane & 15);
    const float g  = gain[col];
    const float bb = bias[col];
#pragma unroll
    for (int im = 0; im < 2; ++im) {
      const int rbase = (mb0 + im) * 16 + ((lane < 16) ? 0 : 8);
      const v8f c = acc[im][jn];
#pragma unroll
      for (int r = 0; r < 8; ++r) {
        const float val = tanhf(c[r] * inv_k) * g + bb;
        out[(size_t)(rbase + r) * HDIM + col] = val;
      }
    }
  }
}

// ---------------------------------------------------------------------------
// Host-side launcher. Inputs (setup_inputs order):
//   0:t  1:h[2048,1024]  2:W[1024,1024,16]  3:alpha[16]  4:beta[16]
//   5:gain[1024]  6:bias[1024]     out: f32 [2048,1024]
// Workspace: [0, 32 MB) W bf16 B-fragments, [32 MB, 96 MB) S bf16 A-fragments.
// ---------------------------------------------------------------------------
extern "C" void kernel_launch(void* const* d_in, const int* in_sizes, int n_in,
                              void* d_out, int out_size, void* d_ws, size_t ws_size,
                              hipStream_t stream) {
  const float* h     = (const float*)d_in[1];
  const float* W     = (const float*)d_in[2];
  const float* alpha = (const float*)d_in[3];
  const float* beta  = (const float*)d_in[4];
  const float* gain  = (const float*)d_in[5];
  const float* bias  = (const float*)d_in[6];
  float* out = (float*)d_out;

  unsigned short* wfrag = (unsigned short*)d_ws;                       // 16.78M ushort = 32 MB
  unsigned short* sfrag = wfrag + (size_t)NB_FRAGS * KB_STEPS * 512;   // 33.55M ushort = 64 MB

  {
    const int total = NB_FRAGS * KB_STEPS * 32;     // 1,048,576 threads
    prep_w_frags<<<total / 256, 256, 0, stream>>>(W, wfrag);
  }
  {
    const int total = MB_FRAGS * KB_STEPS * 32;     // 2,097,152 threads
    prep_s_frags<<<total / 256, 256, 0, stream>>>(h, alpha, beta, sfrag);
  }
  {
    dim3 grid(HDIM / 128, BSZ / 128);               // (8, 16) = 128 workgroups
    kan_wmma_gemm<<<grid, 256, 0, stream>>>(sfrag, wfrag, gain, bias, out);
  }
}
